// GATLayer_3556232922574
// MI455X (gfx1250) — matrix operand order repaired and checked by reading, not compiled, not run on hardware
//
#include <hip/hip_runtime.h>
#include <hip/hip_bf16.h>

#define DIM 128
#define LEAKY 0.01f

typedef __bf16 v16bf __attribute__((ext_vector_type(16)));
typedef __bf16 v8bf  __attribute__((ext_vector_type(8)));
typedef __bf16 v4bf  __attribute__((ext_vector_type(4)));
typedef float  v8f   __attribute__((ext_vector_type(8)));

// ---- helpers -------------------------------------------------------------

__device__ __forceinline__ __bf16 f2bf(float f) {
    unsigned u = __float_as_uint(f);
    unsigned r = (u + 0x7FFFu + ((u >> 16) & 1u)) >> 16;   // RNE truncate
    unsigned short s = (unsigned short)r;
    return __builtin_bit_cast(__bf16, s);
}

// order-preserving float <-> uint map for atomicMax on signed floats
__device__ __forceinline__ unsigned fenc(float f) {
    unsigned u = __float_as_uint(f);
    return (u & 0x80000000u) ? ~u : (u | 0x80000000u);
}
__device__ __forceinline__ float fdec(unsigned u) {
    return __uint_as_float((u & 0x80000000u) ? (u ^ 0x80000000u) : ~u);
}

__device__ __forceinline__ v16bf cat8(v8bf lo, v8bf hi) {
    return __builtin_shufflevector(lo, hi, 0, 1, 2, 3, 4, 5, 6, 7,
                                           8, 9, 10, 11, 12, 13, 14, 15);
}

// ---- zero scratch --------------------------------------------------------

__global__ void zero_f(float* __restrict__ p, long long n) {
    long long i = (long long)blockIdx.x * blockDim.x + threadIdx.x;
    long long stride = (long long)gridDim.x * blockDim.x;
    for (; i < n; i += stride) p[i] = 0.0f;
}

// ---- one-shot fp32 -> bf16 conversion (vectorized, memory bound) ---------

__global__ void cvt_bf16(const float4* __restrict__ s, v4bf* __restrict__ d,
                         long long n4) {
    long long i = (long long)blockIdx.x * blockDim.x + threadIdx.x;
    long long stride = (long long)gridDim.x * blockDim.x;
    for (; i < n4; i += stride) {
        float4 f = s[i];
        v4bf o;
        o[0] = f2bf(f.x); o[1] = f2bf(f.y); o[2] = f2bf(f.z); o[3] = f2bf(f.w);
        d[i] = o;
    }
}

// ---- shared WMMA GEMM core ----------------------------------------------
// One wave per 16-row tile; 8 col-tiles of 16; K=128 in 4 chunks of 32.
// A fragment (16x32 bf16): lane holds row l16; K-half by lane[4]:
//   K = half*16 + e (16 contiguous bf16).
// B fragment (32x16 bf16): col = l16, K = half*16 + e (16 contiguous bf16).
// All 8 B fragments of a K-chunk are loaded first (one clause, one wait),
// then 8 WMMAs issue back-to-back.

__device__ __forceinline__ void gemm_tile(
        const __bf16* __restrict__ A, const __bf16* __restrict__ W,
        size_t row, int half, int l16, v8f acc[8]) {
#pragma unroll
    for (int kc = 0; kc < 4; ++kc) {
        const __bf16* ap = A + row * DIM + kc * 32 + half * 16;
        v16bf a = cat8(*(const v8bf*)ap, *(const v8bf*)(ap + 8));
        v16bf b[8];
#pragma unroll
        for (int ct = 0; ct < 8; ++ct) {
            const __bf16* bp = W + (size_t)(ct * 16 + l16) * DIM + kc * 32 + half * 16;
            b[ct] = cat8(*(const v8bf*)bp, *(const v8bf*)(bp + 8));
        }
#pragma unroll
        for (int ct = 0; ct < 8; ++ct)
            acc[ct] = __builtin_amdgcn_wmma_f32_16x16x32_bf16(
                false, a, false, b[ct], (short)0, acc[ct], false, false);
    }
}

// ---- WMMA GEMM: messages = in_states @ W_msg^T ---------------------------

__global__ __launch_bounds__(256) void gemm_msg(
        const __bf16* __restrict__ A,   // [N,128] bf16
        const __bf16* __restrict__ W,   // [128,128] bf16 (row = out, col = in)
        float* __restrict__ Mout,       // [N,128] messages fp32
        int nTiles) {
    int wave = (int)((blockIdx.x * blockDim.x + threadIdx.x) >> 5);
    if (wave >= nTiles) return;                      // wave-uniform guard
    int lane = threadIdx.x & 31;
    int half = lane >> 4;
    int l16  = lane & 15;
    size_t row = (size_t)wave * 16 + l16;

    v8f acc[8];
#pragma unroll
    for (int c = 0; c < 8; ++c)
#pragma unroll
        for (int g = 0; g < 8; ++g) acc[c][g] = 0.0f;

    gemm_tile(A, W, row, half, l16, acc);

    // D layout: row = half*8 + g, col = l16
#pragma unroll
    for (int ct = 0; ct < 8; ++ct)
#pragma unroll
        for (int g = 0; g < 8; ++g)
            Mout[((size_t)wave * 16 + half * 8 + g) * DIM + ct * 16 + l16] = acc[ct][g];
}

// ---- per-node attention dot products ------------------------------------

__global__ __launch_bounds__(256) void node_dots(
        const float* __restrict__ M, const float* __restrict__ attn,
        float* __restrict__ sf, float* __restrict__ st, int n) {
    int node = (int)((blockIdx.x * blockDim.x + threadIdx.x) >> 5);
    if (node >= n) return;
    int lane = threadIdx.x & 31;
    float4 m  = ((const float4*)(M + (size_t)node * DIM))[lane];
    float4 w0 = ((const float4*)attn)[lane];
    float4 w1 = ((const float4*)(attn + DIM))[lane];
    float f = m.x * w0.x + m.y * w0.y + m.z * w0.z + m.w * w0.w;
    float t = m.x * w1.x + m.y * w1.y + m.z * w1.z + m.w * w1.w;
#pragma unroll
    for (int off = 16; off; off >>= 1) {
        f += __shfl_down(f, off, 32);
        t += __shfl_down(t, off, 32);
    }
    if (lane == 0) { sf[node] = f; st[node] = t; }
}

// ---- edge pass 1: LeakyReLU score + segment max --------------------------

__global__ void edge_scores(const int* __restrict__ src, const int* __restrict__ dst,
                            const float* __restrict__ sf, const float* __restrict__ st,
                            float* __restrict__ eraw, unsigned* __restrict__ segmax,
                            int E) {
    int i = blockIdx.x * blockDim.x + threadIdx.x;
    if (i >= E) return;
    float e = sf[src[i]] + st[dst[i]];
    e = (e >= 0.0f) ? e : LEAKY * e;
    eraw[i] = e;
    atomicMax(&segmax[dst[i]], fenc(e));
}

// ---- edge pass 2: exp + segment sum --------------------------------------

__global__ void edge_exp(const int* __restrict__ dst,
                         const float* __restrict__ eraw, const unsigned* __restrict__ segmax,
                         float* __restrict__ eexp, float* __restrict__ segsum, int E) {
    int i = blockIdx.x * blockDim.x + threadIdx.x;
    if (i >= E) return;
    int d = dst[i];
    float x = __expf(eraw[i] - fdec(segmax[d]));
    eexp[i] = x;
    atomicAdd(&segsum[d], x);
}

// ---- edge pass 3: weighted scatter-add (one wave per edge) ---------------

__global__ __launch_bounds__(256) void scatter_msgs(
        const int* __restrict__ src, const int* __restrict__ dst,
        const float* __restrict__ eexp, const float* __restrict__ segsum,
        const float* __restrict__ M, float* __restrict__ merged, int E) {
    int edge = (int)((blockIdx.x * blockDim.x + threadIdx.x) >> 5);
    if (edge >= E) return;
    int lane = threadIdx.x & 31;
    int s = src[edge], d = dst[edge];
    float alpha = eexp[edge] / segsum[d];
    float4 m = ((const float4*)(M + (size_t)s * DIM))[lane];
    float* out = merged + (size_t)d * DIM + lane * 4;
    atomicAdd(out + 0, alpha * m.x);
    atomicAdd(out + 1, alpha * m.y);
    atomicAdd(out + 2, alpha * m.z);
    atomicAdd(out + 3, alpha * m.w);
}

// ---- WMMA GEMM + epilogue: out = relu(merged) + sig * (in @ W_res^T) -----

__global__ __launch_bounds__(256) void gemm_res(
        const __bf16* __restrict__ A,   // [N,128] bf16 in_states
        const __bf16* __restrict__ W,   // [128,128] bf16 W_res
        const float* __restrict__ merged,
        const float* __restrict__ coef, // 1 element
        float* __restrict__ out, int nTiles) {
    int wave = (int)((blockIdx.x * blockDim.x + threadIdx.x) >> 5);
    if (wave >= nTiles) return;
    int lane = threadIdx.x & 31;
    int half = lane >> 4;
    int l16  = lane & 15;
    size_t row = (size_t)wave * 16 + l16;
    float sig = 1.0f / (1.0f + __expf(-coef[0]));

    v8f acc[8];
#pragma unroll
    for (int c = 0; c < 8; ++c)
#pragma unroll
        for (int g = 0; g < 8; ++g) acc[c][g] = 0.0f;

    gemm_tile(A, W, row, half, l16, acc);

    // epilogue: relu(merged) + sigmoid * GEMM result
#pragma unroll
    for (int ct = 0; ct < 8; ++ct)
#pragma unroll
        for (int g = 0; g < 8; ++g) {
            size_t idx = ((size_t)wave * 16 + half * 8 + g) * DIM + ct * 16 + l16;
            float v = merged[idx];
            v = v > 0.0f ? v : 0.0f;
            out[idx] = v + sig * acc[ct][g];
        }
}

// ---- launcher ------------------------------------------------------------

extern "C" void kernel_launch(void* const* d_in, const int* in_sizes, int n_in,
                              void* d_out, int out_size, void* d_ws, size_t ws_size,
                              hipStream_t stream) {
    const float* in_states = (const float*)d_in[0];
    const int*   edges     = (const int*)d_in[1];
    const float* W_msg     = (const float*)d_in[2];
    const float* attn_w    = (const float*)d_in[3];
    const float* W_res     = (const float*)d_in[4];
    const float* coef      = (const float*)d_in[5];
    float* out = (float*)d_out;

    const int n = in_sizes[0] / DIM;       // 100000 nodes
    const int E = in_sizes[1] / 2;         // 1600000 edges
    const int* src = edges;
    const int* dst = edges + E;

    // workspace layout
    float*    messages = (float*)d_ws;                       // n*128 f32
    float*    merged   = messages + (size_t)n * DIM;         // n*128 f32
    float*    sfrom    = merged + (size_t)n * DIM;           // n
    float*    sto      = sfrom + n;                          // n
    float*    segsum   = sto + n;                            // n
    unsigned* segmax   = (unsigned*)(segsum + n);            // n (0u == -inf sentinel)
    float*    eraw     = (float*)(segmax + n);               // E
    float*    eexp     = eraw + E;                           // E
    __bf16*   abf      = (__bf16*)(eexp + E);                // n*128 bf16
    __bf16*   wmbf     = abf + (size_t)n * DIM;              // 128*128 bf16
    __bf16*   wrbf     = wmbf + DIM * DIM;                   // 128*128 bf16

    const int nTiles = (n + 15) / 16;      // 6250 (exact)

    // 1. zero accumulated region: merged, sfrom, sto, segsum, segmax (contiguous)
    long long zcount = (long long)n * DIM + 4LL * n;
    zero_f<<<2048, 256, 0, stream>>>(merged, zcount);

    // 2. pack fp32 -> bf16 once (A and both weight matrices)
    cvt_bf16<<<2048, 256, 0, stream>>>((const float4*)in_states, (v4bf*)abf,
                                       (long long)n * DIM / 4);
    cvt_bf16<<<32, 256, 0, stream>>>((const float4*)W_msg, (v4bf*)wmbf,
                                     (long long)DIM * DIM / 4);
    cvt_bf16<<<32, 256, 0, stream>>>((const float4*)W_res, (v4bf*)wrbf,
                                     (long long)DIM * DIM / 4);

    // 3. messages = in_states @ W_msg^T   (WMMA bf16, f32 accumulate)
    gemm_msg<<<(nTiles + 7) / 8, 256, 0, stream>>>(abf, wmbf, messages, nTiles);

    // 4. per-node attention dots
    node_dots<<<(n + 7) / 8, 256, 0, stream>>>(messages, attn_w, sfrom, sto, n);

    // 5. edge scores + segment max
    edge_scores<<<(E + 255) / 256, 256, 0, stream>>>(src, dst, sfrom, sto, eraw, segmax, E);

    // 6. exp + segment sum
    edge_exp<<<(E + 255) / 256, 256, 0, stream>>>(dst, eraw, segmax, eexp, segsum, E);

    // 7. weighted scatter-add (one wave per edge, L2-resident)
    scatter_msgs<<<(E + 7) / 8, 256, 0, stream>>>(src, dst, eexp, segsum, messages, merged, E);

    // 8. out = relu(merged) + sigmoid(c) * (in_states @ W_res^T)
    gemm_res<<<(nTiles + 7) / 8, 256, 0, stream>>>(abf, wrbf, merged, coef, out, nTiles);
}
